// HierLayer_80728205295744
// MI455X (gfx1250) — matrix operand order repaired and checked by reading, not compiled, not run on hardware
//
#include <hip/hip_runtime.h>
#include <hip/hip_bf16.h>

typedef __attribute__((ext_vector_type(16))) _Float16 v16h;
typedef __attribute__((ext_vector_type(8)))  float    v8f;

#define NEG_SLOPE 0.2f

// ---------------------------------------------------------------------------
// Fused GEMM + attention dots:
//   M  = X (n x 64) @ W (64 x 64)           (f16 WMMA, f32 accumulate)
//   dt = M . a[0:64],  db = M . a[64:128]   (computed from WMMA accumulators)
// One wave computes a 16x64 output tile: 2 k-steps x 4 n-tiles = 8 WMMAs.
// W is staged TRANSPOSED in LDS so each B fragment is one contiguous 32B load.
// ---------------------------------------------------------------------------
__launch_bounds__(256)
__global__ void gemm64_wmma(const float* __restrict__ X, const float* __restrict__ W,
                            const float* __restrict__ a, float* __restrict__ M,
                            float* __restrict__ dt, float* __restrict__ db, int n) {
  __shared__ _Float16 WlT[64 * 64];         // WlT[col*64 + k] = W[k*64 + col]
  for (int i = threadIdx.x; i < 64 * 64; i += 256) {
    const int k = i >> 6, c = i & 63;
    WlT[c * 64 + k] = (_Float16)W[i];       // coalesced global read
  }
  __syncthreads();

  const int wave = threadIdx.x >> 5;
  const int lane = threadIdx.x & 31;
  const int rowTile = (blockIdx.x * 8 + wave) * 16;
  if (rowTile >= n) return;                 // wave-uniform

  const int lrow  = lane & 15;
  const int khalf = lane >> 4;              // 0 or 1
  int r = rowTile + lrow;
  if (r > n - 1) r = n - 1;                 // clamp (keeps EXEC all-ones; pad rows never stored)
  const float* xp = X + (size_t)r * 64;

  v8f acc[4];
#pragma unroll
  for (int t = 0; t < 4; ++t)
#pragma unroll
    for (int j = 0; j < 8; ++j) acc[t][j] = 0.0f;

#pragma unroll
  for (int ks = 0; ks < 2; ++ks) {
    // A fragment (16x32 f16): lane holds row lrow, K = kb..kb+15
    const int kb = ks * 32 + khalf * 16;
    v16h afrag;
#pragma unroll
    for (int j = 0; j < 8; ++j) {
      afrag[j]     = (_Float16)xp[kb + j];
      afrag[8 + j] = (_Float16)xp[kb + 8 + j];
    }
    const int kb2 = ks * 32 + khalf * 16;
#pragma unroll
    for (int t = 0; t < 4; ++t) {
      // B fragment (32x16 f16): lane holds col, K = kb2..kb2+15 -> contiguous in WlT
      const int col = t * 16 + lrow;
      const v16h bfrag = *reinterpret_cast<const v16h*>(&WlT[col * 64 + kb2]);
      acc[t] = __builtin_amdgcn_wmma_f32_16x16x32_f16(false, afrag, false, bfrag,
                                                      (short)0, acc[t], false, false);
    }
  }

  // attention-dot coefficients for this lane's column within each n-tile
  float at_[4], ab_[4];
#pragma unroll
  for (int t = 0; t < 4; ++t) {
    at_[t] = a[t * 16 + lrow];
    ab_[t] = a[64 + t * 16 + lrow];
  }

  // C/D layout: reg v, lanes 0-15 -> row rowTile+v, lanes 16-31 -> row rowTile+8+v
#pragma unroll
  for (int v = 0; v < 8; ++v) {
    const int orow = rowTile + khalf * 8 + v;
    float st = 0.0f, sb = 0.0f;
#pragma unroll
    for (int t = 0; t < 4; ++t) {
      const float av = acc[t][v];
      st += av * at_[t];
      sb += av * ab_[t];
      if (orow < n) M[(size_t)orow * 64 + t * 16 + lrow] = av;
    }
    // reduce across the 16 lanes of this half-wave (full 64-col dot)
#pragma unroll
    for (int mask = 1; mask < 16; mask <<= 1) {
      st += __shfl_xor(st, mask);
      sb += __shfl_xor(sb, mask);
    }
    if (lrow == 0 && orow < n) { dt[orow] = st; db[orow] = sb; }
  }
}

// ---------------------------------------------------------------------------
// Order-preserving float <-> uint mapping for atomicMax on floats
// ---------------------------------------------------------------------------
__device__ __forceinline__ unsigned ordf(float f) {
  unsigned u = __float_as_uint(f);
  return (u & 0x80000000u) ? ~u : (u | 0x80000000u);
}
__device__ __forceinline__ float deord(unsigned u) {
  return (u & 0x80000000u) ? __uint_as_float(u & 0x7fffffffu) : __uint_as_float(~u);
}

// pass 1: e = leaky(sA[seg] + sB[oth]); segment max via atomicMax
__global__ void edge_logit(const int* __restrict__ seg, const int* __restrict__ oth,
                           const float* __restrict__ sA, const float* __restrict__ sB,
                           float* __restrict__ ebuf, unsigned* __restrict__ rmax, int nE) {
  const int i = blockIdx.x * blockDim.x + threadIdx.x;
  if (i >= nE) return;
  const int s = seg[i];
  float v = sA[s] + sB[oth[i]];
  v = v > 0.0f ? v : NEG_SLOPE * v;
  ebuf[i] = v;
  atomicMax(&rmax[s], ordf(v));
}

// pass 2: w = exp(e - max[seg]); segment sum via atomicAdd
__global__ void edge_exp(const int* __restrict__ seg, const float* __restrict__ ebuf,
                         const unsigned* __restrict__ rmax, float* __restrict__ wbuf,
                         float* __restrict__ denom, int nE) {
  const int i = blockIdx.x * blockDim.x + threadIdx.x;
  if (i >= nE) return;
  const int s = seg[i];
  const float w = __expf(ebuf[i] - deord(rmax[s]));
  wbuf[i] = w;
  atomicAdd(&denom[s], w);
}

// pass 3: out[seg] += (w/denom[seg]) * V[oth]  (one wave per edge, 2 floats/lane)
__global__ void edge_scatter(const int* __restrict__ seg, const int* __restrict__ oth,
                             const float* __restrict__ wbuf, const float* __restrict__ denom,
                             const float* __restrict__ V, float* __restrict__ out, int nE) {
  const int t    = blockIdx.x * blockDim.x + threadIdx.x;
  const int i    = t >> 5;
  const int lane = threadIdx.x & 31;
  if (i >= nE) return;
  const int s = seg[i], o = oth[i];
  const float att = wbuf[i] / denom[s];
  const float* vp = V + (size_t)o * 64;
  __builtin_prefetch(vp, 0, 0);
  float* op = out + (size_t)s * 64;
  atomicAdd(&op[lane],      att * vp[lane]);
  atomicAdd(&op[lane + 32], att * vp[lane + 32]);
}

__global__ void k_zero(float* __restrict__ p, long long cnt) {
  const long long i = (long long)blockIdx.x * blockDim.x + threadIdx.x;
  if (i < cnt) p[i] = 0.0f;
}
__global__ void k_scale(float* __restrict__ p, long long cnt, float s) {
  const long long i = (long long)blockIdx.x * blockDim.x + threadIdx.x;
  if (i < cnt) p[i] *= s;
}

// ---------------------------------------------------------------------------
// Host orchestration
// ---------------------------------------------------------------------------
static inline unsigned cdivu(long long a, int b) { return (unsigned)((a + b - 1) / b); }

struct PHbs  { const float *W, *a; };
struct PHbns { const float *Ws, *Wt, *a; };

extern "C" void kernel_launch(void* const* d_in, const int* in_sizes, int n_in,
                              void* d_out, int out_size, void* d_ws, size_t ws_size,
                              hipStream_t stream) {
  if (n_in < 61) return;
  auto F = [&](int i) { return (const float*)d_in[i]; };
  auto I = [&](int i) { return (const int*)d_in[i]; };

  const int n0 = in_sizes[0] / 64, n1 = in_sizes[1] / 64, n2 = in_sizes[2] / 64,
            n3 = in_sizes[3] / 64, n4 = in_sizes[4] / 64;
  const float *x0 = F(0), *x1 = F(1), *x2 = F(2), *x3 = F(3), *x4 = F(4);

  const int EA1 = in_sizes[6] / 2, EA2 = in_sizes[7] / 2, EA3 = in_sizes[8] / 2, EA4 = in_sizes[9] / 2;
  const int EI01 = in_sizes[10] / 2, EI12 = in_sizes[14] / 2, EI23 = in_sizes[17] / 2, EI34 = in_sizes[19] / 2;
  const int *A1r = I(6),  *A1c = I(6) + EA1;
  const int *A2r = I(7),  *A2c = I(7) + EA2;
  const int *A3r = I(8),  *A3c = I(8) + EA3;
  const int *A4r = I(9),  *A4c = I(9) + EA4;
  const int *I01r = I(10), *I01c = I(10) + EI01;
  const int *I12r = I(14), *I12c = I(14) + EI12;
  const int *I23r = I(17), *I23c = I(17) + EI23;
  const int *I34r = I(19), *I34c = I(19) + EI34;

  // ---- params (handle both insertion-order and JAX-sorted flattening) ----
  auto hb = [&](int base) { return PHbs{F(base), F(base + 1)}; };
  auto hn = [&](int base) { return PHbns{F(base), F(base + 1), F(base + 2)}; };
  PHbs  hbs_1_l1, hbs_2_l1, hbs_3_l1, hbs_4_l1, hbs_2_l2, hbs_3_l2, hbs_3_l3;
  PHbns hbns_0_1_l1, hbns_1_2_l1, hbns_2_3_l1, hbns_3_4_l1, hbns_1_2_l2,
        hbns_2_3_l2, hbns_3_4_l2, hbns_2_3_l3, hbns_3_4_l3;
  if (in_sizes[21] == 128) {  // insertion order: 7x hbs(W,a) then 9x hbns(Ws,Wt,a)
    hbs_1_l1 = hb(20); hbs_2_l1 = hb(22); hbs_3_l1 = hb(24); hbs_4_l1 = hb(26);
    hbs_2_l2 = hb(28); hbs_3_l2 = hb(30); hbs_3_l3 = hb(32);
    hbns_0_1_l1 = hn(34); hbns_1_2_l1 = hn(37); hbns_2_3_l1 = hn(40); hbns_3_4_l1 = hn(43);
    hbns_1_2_l2 = hn(46); hbns_2_3_l2 = hn(49); hbns_3_4_l2 = hn(52);
    hbns_2_3_l3 = hn(55); hbns_3_4_l3 = hn(58);
  } else {                    // sorted key order: 9x hbns then 7x hbs
    hbns_0_1_l1 = hn(20); hbns_1_2_l1 = hn(23); hbns_1_2_l2 = hn(26);
    hbns_2_3_l1 = hn(29); hbns_2_3_l2 = hn(32); hbns_2_3_l3 = hn(35);
    hbns_3_4_l1 = hn(38); hbns_3_4_l2 = hn(41); hbns_3_4_l3 = hn(44);
    hbs_1_l1 = hb(47); hbs_2_l1 = hb(49); hbs_2_l2 = hb(51);
    hbs_3_l1 = hb(53); hbs_3_l2 = hb(55); hbs_3_l3 = hb(57); hbs_4_l1 = hb(59);
  }

  // ---- workspace bump allocator ----
  int NMAX = n0; if (n1 > NMAX) NMAX = n1; if (n2 > NMAX) NMAX = n2;
  if (n3 > NMAX) NMAX = n3; if (n4 > NMAX) NMAX = n4;
  int EMAX = EA1; if (EA2 > EMAX) EMAX = EA2; if (EI12 > EMAX) EMAX = EI12;
  if (EI01 > EMAX) EMAX = EI01; if (EA3 > EMAX) EMAX = EA3; if (EI23 > EMAX) EMAX = EI23;

  char* wp = (char*)d_ws;
  auto walloc = [&](size_t bytes) -> char* {
    char* p = wp; wp += (bytes + 255) & ~(size_t)255; return p;
  };
  float*    MS    = (float*)walloc((size_t)NMAX * 64 * 4);
  float*    MT    = (float*)walloc((size_t)NMAX * 64 * 4);
  float*    dtS   = (float*)walloc((size_t)NMAX * 4);
  float*    dbS   = (float*)walloc((size_t)NMAX * 4);
  float*    dtT   = (float*)walloc((size_t)NMAX * 4);
  float*    dbT   = (float*)walloc((size_t)NMAX * 4);
  float*    X2L1  = (float*)walloc((size_t)n2 * 64 * 4);
  float*    X3L1  = (float*)walloc((size_t)n3 * 64 * 4);
  float*    X3L2  = (float*)walloc((size_t)n3 * 64 * 4);
  float*    ebuf  = (float*)walloc((size_t)EMAX * 4);
  float*    wbuf  = (float*)walloc((size_t)EMAX * 4);
  unsigned* rmax  = (unsigned*)walloc((size_t)NMAX * 4);
  float*    denom = (float*)walloc((size_t)NMAX * 4);

  // ---- launch helpers ----
  auto zero = [&](float* p, long long cnt) {
    k_zero<<<cdivu(cnt, 256), 256, 0, stream>>>(p, cnt);
  };
  auto scale = [&](float* p, long long cnt, float s) {
    k_scale<<<cdivu(cnt, 256), 256, 0, stream>>>(p, cnt, s);
  };
  auto gemm = [&](const float* X, const float* W, const float* a,
                  float* M, float* dt, float* db, int n) {
    gemm64_wmma<<<cdivu(n, 128), 256, 0, stream>>>(X, W, a, M, dt, db, n);
  };
  auto attn = [&](const int* seg, const int* oth, const float* sA, const float* sB,
                  const float* V, float* out, int nSeg, int nE) {
    zero((float*)rmax, nSeg);
    zero(denom, nSeg);
    edge_logit<<<cdivu(nE, 256), 256, 0, stream>>>(seg, oth, sA, sB, ebuf, rmax, nE);
    edge_exp<<<cdivu(nE, 256), 256, 0, stream>>>(seg, ebuf, rmax, wbuf, denom, nE);
    edge_scatter<<<cdivu((long long)nE * 32, 256), 256, 0, stream>>>(seg, oth, wbuf, denom, V, out, nE);
  };
  // hbs: e = leaky(p[row]+q[col]) over rows; out[row] += att*m[col]
  auto run_hbs = [&](const float* X, PHbs P, const int* rows, const int* cols,
                     int n, int nE, float* out) {
    gemm(X, P.W, P.a, MS, dtS, dbS, n);
    attn(rows, cols, dtS, dbS, MS, out, n, nE);
  };
  // hbns msg_on_source: f = leaky(pt[row]+qs[col]), softmax over cols; out[col] += f*t_msg[row]
  auto run_hbns_src = [&](const float* Xs, const float* Xt, PHbns P,
                          const int* rows, const int* cols, int ns, int nt, int nE, float* out) {
    gemm(Xs, P.Ws, P.a, MS, dtS, dbS, ns);
    gemm(Xt, P.Wt, P.a, MT, dtT, dbT, nt);
    attn(cols, rows, dbS, dtT, MT, out, ns, nE);
  };
  // hbns msg_on_target: e = leaky(ps[col]+qt[row]), softmax over rows; out[row] += e*s_msg[col]
  auto run_hbns_tgt = [&](const float* Xs, const float* Xt, PHbns P,
                          const int* rows, const int* cols, int ns, int nt, int nE, float* out) {
    gemm(Xs, P.Ws, P.a, MS, dtS, dbS, ns);
    gemm(Xt, P.Wt, P.a, MT, dtT, dbT, nt);
    attn(rows, cols, dbT, dtS, MS, out, nt, nE);
  };

  // ---- output regions ----
  float* out0 = (float*)d_out;
  float* out1 = out0 + (size_t)n0 * 64;
  float* out2 = out1 + (size_t)n1 * 64;
  float* out3 = out2 + (size_t)n2 * 64;
  float* out4 = out3 + (size_t)n3 * 64;

  hipMemcpyAsync(out0, x0, (size_t)n0 * 64 * sizeof(float), hipMemcpyDeviceToDevice, stream);

  // ---------------- Level 1 ----------------
  zero(out1, (long long)n1 * 64);
  run_hbs(x1, hbs_1_l1, A1r, A1c, n1, EA1, out1);
  run_hbns_src(x1, x0, hbns_0_1_l1, I01r, I01c, n1, n0, EI01, out1);
  scale(out1, (long long)n1 * 64, 0.5f);

  zero(X2L1, (long long)n2 * 64);
  run_hbs(x2, hbs_2_l1, A2r, A2c, n2, EA2, X2L1);
  run_hbns_src(x2, x1, hbns_1_2_l1, I12r, I12c, n2, n1, EI12, X2L1);
  scale(X2L1, (long long)n2 * 64, 0.5f);

  zero(X3L1, (long long)n3 * 64);
  run_hbs(x3, hbs_3_l1, A3r, A3c, n3, EA3, X3L1);
  run_hbns_src(x3, x2, hbns_2_3_l1, I23r, I23c, n3, n2, EI23, X3L1);
  scale(X3L1, (long long)n3 * 64, 0.5f);

  zero(out4, (long long)n4 * 64);     // x_4_l1 == x_4_l2 (final)
  run_hbs(x4, hbs_4_l1, A4r, A4c, n4, EA4, out4);
  run_hbns_src(x4, x3, hbns_3_4_l1, I34r, I34c, n4, n3, EI34, out4);
  scale(out4, (long long)n4 * 64, 0.5f);

  // ---------------- Level 2 ----------------
  zero(out2, (long long)n2 * 64);
  run_hbs(X2L1, hbs_2_l2, A2r, A2c, n2, EA2, out2);
  run_hbns_src(X2L1, out1, hbns_1_2_l2, I12r, I12c, n2, n1, EI12, out2);
  scale(out2, (long long)n2 * 64, 0.5f);

  zero(X3L2, (long long)n3 * 64);
  run_hbs(X3L1, hbs_3_l2, A3r, A3c, n3, EA3, X3L2);
  run_hbns_src(X3L1, X2L1, hbns_2_3_l2, I23r, I23c, n3, n2, EI23, X3L2);
  run_hbns_tgt(out4, X3L1, hbns_3_4_l2, I34r, I34c, n4, n3, EI34, X3L2);
  scale(X3L2, (long long)n3 * 64, 1.0f / 3.0f);

  // ---------------- Level 3 ----------------
  zero(out3, (long long)n3 * 64);
  run_hbs(X3L2, hbs_3_l3, A3r, A3c, n3, EA3, out3);
  run_hbns_src(X3L2, out2, hbns_2_3_l3, I23r, I23c, n3, n2, EI23, out3);
  run_hbns_tgt(out4, X3L2, hbns_3_4_l3, I34r, I34c, n4, n3, EI34, out3);
  scale(out3, (long long)n3 * 64, 1.0f / 3.0f);

  (void)out_size; (void)ws_size;
}
